// SocialLSTM_33268816674870
// MI455X (gfx1250) — matrix operand order repaired and checked by reading, not compiled, not run on hardware
//
#include <hip/hip_runtime.h>
#include <math.h>

// Social LSTM for MI455X (gfx1250, wave32, WMMA).
// Roofline: the fp32 `grids` tensor (335 MB total, 16.8 MB/step) streams from HBM once
// -> ~0.72 us/step floor at 23.3 TB/s; all GEMM compute fits far under that with
// v_wmma_f32_16x16x32_bf16. The big einsum is reassociated as
//   te = grid[n,(m,g)] @ P[(m,g),e],  P[(m,g),e] = sum_r h_masked[m,r]*W_ten[e,g*128+r]
// so the hot GEMM reads grid in its natural contiguous layout (2x fewer FLOPs than
// materializing `social`). Grid loads use non-temporal hints (one-pass stream; keep
// 192 MB L2 for the 32x-reused PbfT + weights). Weights are converted to bf16 once,
// outside the 20-step sequential loop.

#define SEQ   20
#define NN    512
#define G2    16
#define RR    128
#define EE    64
#define OUTD  5
#define KTEN  (NN * G2)   // 8192
#define XCATK 256         // [ie(64) | te(64) | h(128)]

typedef __attribute__((ext_vector_type(16))) __bf16 v16bf;
typedef __attribute__((ext_vector_type(8)))  float  v8f;
typedef __attribute__((ext_vector_type(4)))  float  v4f;

static __device__ __forceinline__ __bf16 f2bf(float x) { return (__bf16)x; }

// ---------------- CDNA5 wave32 WMMA fragment helpers (CDNA5 ISA 7.12.2) ----------
// A 16x32 (MxK) bf16: lane L -> m = m0+(L&15), hi = L>>4;
// half i (i<8): K = k0 + hi*8 + i ; half i (i>=8): K = k0 + 16 + hi*8 + (i-8)
static __device__ __forceinline__ v16bf load_a_bf16(const __bf16* __restrict__ src, int ld,
                                                    int m0, int k0, int lane) {
  const __bf16* row = src + (size_t)(m0 + (lane & 15)) * ld + k0 + ((lane >> 4) << 3);
  v16bf a;
#pragma unroll
  for (int j = 0; j < 8; ++j) { a[j] = row[j]; a[8 + j] = row[16 + j]; }
  return a;
}
// A fragment from fp32 source with non-temporal (streaming) loads + cvt to bf16
static __device__ __forceinline__ v16bf load_a_f32_nt(const float* __restrict__ src, int ld,
                                                      int m0, int k0, int lane) {
  const float* row = src + (size_t)(m0 + (lane & 15)) * ld + k0 + ((lane >> 4) << 3);
  v4f r0 = __builtin_nontemporal_load((const v4f*)(row));
  v4f r1 = __builtin_nontemporal_load((const v4f*)(row + 4));
  v4f r2 = __builtin_nontemporal_load((const v4f*)(row + 16));
  v4f r3 = __builtin_nontemporal_load((const v4f*)(row + 20));
  v16bf a;
#pragma unroll
  for (int j = 0; j < 4; ++j) {
    a[j]      = f2bf(r0[j]);
    a[4 + j]  = f2bf(r1[j]);
    a[8 + j]  = f2bf(r2[j]);
    a[12 + j] = f2bf(r3[j]);
  }
  return a;
}
// B 32x16 (KxN) bf16, element (k,n) = src[n*ld + k] (k contiguous per n).
// lane L -> n = n0+(L&15), hi = L>>4; half i (i<8): K = k0 + hi*8 + i ; half i (i>=8): K = k0 + 16 + hi*8 + (i-8)
static __device__ __forceinline__ v16bf load_b_bf16(const __bf16* __restrict__ src, int ld,
                                                    int n0, int k0, int lane) {
  const __bf16* p = src + (size_t)(n0 + (lane & 15)) * ld + k0 + ((lane >> 4) << 3);
  v16bf b;
#pragma unroll
  for (int j = 0; j < 8; ++j) { b[j] = p[j]; b[8 + j] = p[16 + j]; }
  return b;
}
// D/C f32 16x16: VGPR v, lane L -> m = m0 + v + 8*(L>>4), n = n0 + (L&15)
static __device__ __forceinline__ void store_d_f32(float* __restrict__ dst, int ld,
                                                   int m0, int n0, int lane, v8f d) {
  int hi = lane >> 4, nrel = lane & 15;
#pragma unroll
  for (int v = 0; v < 8; ++v)
    dst[(size_t)(m0 + v + 8 * hi) * ld + n0 + nrel] = d[v];
}

#define WMMA_BF16(A, B, C) \
  __builtin_amdgcn_wmma_f32_16x16x32_bf16(false, (A), false, (B), (short)0, (C), false, false)

// ---------------------------------- kernels ------------------------------------------

__global__ void init_state_kernel(const float* __restrict__ h0, const float* __restrict__ c0,
                                  float* __restrict__ h, float* __restrict__ c) {
  int i = blockIdx.x * blockDim.x + threadIdx.x;
  if (i < NN * RR) { h[i] = h0[i]; c[i] = c0[i]; }
}

// One-time (per launch) weight conversion, hoisted out of the sequential t-loop:
//   wten_bf[e][k]  = bf16(W_ten[e][k])                       (64 x 2048)
//   wcat_bf[j][k]  = bf16(k<128 ? W_ih[j][k] : W_hh[j][k-128]) (512 x 256) fused B matrix
__global__ void convert_weights_kernel(const float* __restrict__ W_ten,
                                       const float* __restrict__ W_ih,
                                       const float* __restrict__ W_hh,
                                       __bf16* __restrict__ wten_bf,
                                       __bf16* __restrict__ wcat_bf) {
  int i = blockIdx.x * blockDim.x + threadIdx.x;   // 131072 threads
  if (i >= EE * G2 * RR) return;                   // == 4*RR*XCATK == 131072
  wten_bf[i] = f2bf(W_ten[i]);
  int j = i >> 8, k = i & 255;
  wcat_bf[i] = f2bf(k < 128 ? W_ih[j * RR + k] : W_hh[j * RR + (k - 128)]);
}

// Per-step prep: xcat_bf[:,128:256] = bf16(h); hmask_bf = bf16(h*m); xcat_bf[:,0:64] = bf16(relu(ie))
__global__ void prep_kernel(const float* __restrict__ h, const float* __restrict__ x_t,
                            const int* __restrict__ pres_t,
                            const float* __restrict__ W_in, const float* __restrict__ b_in,
                            __bf16* __restrict__ hmask_bf, __bf16* __restrict__ xcat_bf) {
  int n = blockIdx.x, r = threadIdx.x;
  float m = (float)pres_t[n];
  float hv = h[n * RR + r];
  xcat_bf[n * XCATK + 128 + r] = f2bf(hv);
  hmask_bf[n * RR + r] = f2bf(hv * m);
  if (r < EE) {
    float ie = x_t[n * 2 + 0] * W_in[r * 2 + 0] + x_t[n * 2 + 1] * W_in[r * 2 + 1] + b_in[r];
    xcat_bf[n * XCATK + r] = f2bf(ie > 0.f ? ie : 0.f);
  }
}

// P[(m,g),e] = sum_r hmask[m,r] * wten[e, g*128+r]; stored transposed: PbfT[e][m*16+g] (bf16)
// 512 waves: wave = (mtile:32, g:16); each computes a 16(m) x 64(e) tile, K = 128.
__global__ void p_kernel(const __bf16* __restrict__ hmask_bf,
                         const __bf16* __restrict__ wten_bf, __bf16* __restrict__ PbfT) {
  int tid = blockIdx.x * blockDim.x + threadIdx.x;
  int wid = tid >> 5, lane = tid & 31;
  int mtile = wid >> 4, g = wid & 15;
  int m0 = mtile * 16;
  v8f a0 = {}, a1 = {}, a2 = {}, a3 = {};
  const __bf16* wg = wten_bf + g * RR;   // B(k,e) = wg[e*2048 + k]
#pragma unroll
  for (int kt = 0; kt < RR / 32; ++kt) {
    int k0 = kt * 32;
    v16bf a  = load_a_bf16(hmask_bf, RR, m0, k0, lane);
    v16bf b0 = load_b_bf16(wg, G2 * RR, 0,  k0, lane);
    v16bf b1 = load_b_bf16(wg, G2 * RR, 16, k0, lane);
    v16bf b2 = load_b_bf16(wg, G2 * RR, 32, k0, lane);
    v16bf b3 = load_b_bf16(wg, G2 * RR, 48, k0, lane);
    a0 = WMMA_BF16(a, b0, a0);
    a1 = WMMA_BF16(a, b1, a1);
    a2 = WMMA_BF16(a, b2, a2);
    a3 = WMMA_BF16(a, b3, a3);
  }
  int hi = lane >> 4, nrel = lane & 15;
#pragma unroll
  for (int v = 0; v < 8; ++v) {
    int mk = (m0 + v + 8 * hi) * G2 + g;
    PbfT[(size_t)(0  + nrel) * KTEN + mk] = f2bf(a0[v]);
    PbfT[(size_t)(16 + nrel) * KTEN + mk] = f2bf(a1[v]);
    PbfT[(size_t)(32 + nrel) * KTEN + mk] = f2bf(a2[v]);
    PbfT[(size_t)(48 + nrel) * KTEN + mk] = f2bf(a3[v]);
  }
}

// te_pre partials: grid[n, 0:8192] @ PbfT^T, K split 16 ways for occupancy.
// 512 waves: wave = (mtile:32, s:16); 16(n) x 64(e) tile over K in [s*512,(s+1)*512).
// Grid A-fragments use NT loads (one-pass HBM stream, don't churn L2).
__global__ void te_kernel(const float* __restrict__ grids_t, const __bf16* __restrict__ PbfT,
                          float* __restrict__ tepart) {
  int tid = blockIdx.x * blockDim.x + threadIdx.x;
  int wid = tid >> 5, lane = tid & 31;
  int mtile = wid >> 4, s = wid & 15;
  int m0 = mtile * 16;
  int kbase = s * (KTEN / 16);
  v8f a0 = {}, a1 = {}, a2 = {}, a3 = {};
  for (int kt = 0; kt < (KTEN / 16) / 32; ++kt) {   // 16 k-steps of 32
    int k0 = kbase + kt * 32;
    v16bf a  = load_a_f32_nt(grids_t, KTEN, m0, k0, lane);
    v16bf b0 = load_b_bf16(PbfT, KTEN, 0,  k0, lane);
    v16bf b1 = load_b_bf16(PbfT, KTEN, 16, k0, lane);
    v16bf b2 = load_b_bf16(PbfT, KTEN, 32, k0, lane);
    v16bf b3 = load_b_bf16(PbfT, KTEN, 48, k0, lane);
    a0 = WMMA_BF16(a, b0, a0);
    a1 = WMMA_BF16(a, b1, a1);
    a2 = WMMA_BF16(a, b2, a2);
    a3 = WMMA_BF16(a, b3, a3);
  }
  float* dst = tepart + (size_t)s * (NN * EE);
  store_d_f32(dst, EE, m0, 0,  lane, a0);
  store_d_f32(dst, EE, m0, 16, lane, a1);
  store_d_f32(dst, EE, m0, 32, lane, a2);
  store_d_f32(dst, EE, m0, 48, lane, a3);
}

// Reduce K-split partials, add bias, relu, write bf16 into xcat[:,64:128]
__global__ void te_reduce_kernel(const float* __restrict__ tepart, const float* __restrict__ b_ten,
                                 __bf16* __restrict__ xcat_bf) {
  int idx = blockIdx.x * blockDim.x + threadIdx.x;   // NN*EE
  int n = idx >> 6, e = idx & 63;
  float s = b_ten[e];
#pragma unroll
  for (int p = 0; p < 16; ++p) s += tepart[(size_t)p * (NN * EE) + idx];
  xcat_bf[n * XCATK + EE + e] = f2bf(s > 0.f ? s : 0.f);
}

// gates[n,j] = xcat[n,0:256] @ wcat[j,0:256]^T  (biases added in lstm kernel)
// 256 waves: wave = (mtile:32, jgrp:8); 16(n) x 64(j) tile, K = 256.
__global__ void gates_kernel(const __bf16* __restrict__ xcat_bf,
                             const __bf16* __restrict__ wcat_bf,
                             float* __restrict__ gates) {
  int tid = blockIdx.x * blockDim.x + threadIdx.x;
  int wid = tid >> 5, lane = tid & 31;
  int mtile = wid >> 3, jgrp = wid & 7;
  int m0 = mtile * 16, j0 = jgrp * 64;
  v8f a0 = {}, a1 = {}, a2 = {}, a3 = {};
#pragma unroll
  for (int kt = 0; kt < XCATK / 32; ++kt) {
    int k0 = kt * 32;
    v16bf a  = load_a_bf16(xcat_bf, XCATK, m0, k0, lane);
    v16bf b0 = load_b_bf16(wcat_bf, XCATK, j0 + 0,  k0, lane);
    v16bf b1 = load_b_bf16(wcat_bf, XCATK, j0 + 16, k0, lane);
    v16bf b2 = load_b_bf16(wcat_bf, XCATK, j0 + 32, k0, lane);
    v16bf b3 = load_b_bf16(wcat_bf, XCATK, j0 + 48, k0, lane);
    a0 = WMMA_BF16(a, b0, a0);
    a1 = WMMA_BF16(a, b1, a1);
    a2 = WMMA_BF16(a, b2, a2);
    a3 = WMMA_BF16(a, b3, a3);
  }
  store_d_f32(gates, 4 * RR, m0, j0 + 0,  lane, a0);
  store_d_f32(gates, 4 * RR, m0, j0 + 16, lane, a1);
  store_d_f32(gates, 4 * RR, m0, j0 + 32, lane, a2);
  store_d_f32(gates, 4 * RR, m0, j0 + 48, lane, a3);
}

// LSTM cell elementwise + masked state update (torch gate order i,f,g,o)
__global__ void lstm_kernel(const float* __restrict__ gates,
                            const float* __restrict__ b_ih, const float* __restrict__ b_hh,
                            const int* __restrict__ pres_t,
                            float* __restrict__ h, float* __restrict__ c) {
  int idx = blockIdx.x * blockDim.x + threadIdx.x;   // NN*RR
  int n = idx >> 7, r = idx & 127;
  float m = (float)pres_t[n];
  const float* g = gates + (size_t)n * (4 * RR);
  float gi = g[r]          + b_ih[r]          + b_hh[r];
  float gf = g[RR + r]     + b_ih[RR + r]     + b_hh[RR + r];
  float gg = g[2 * RR + r] + b_ih[2 * RR + r] + b_hh[2 * RR + r];
  float go = g[3 * RR + r] + b_ih[3 * RR + r] + b_hh[3 * RR + r];
  float si = 1.f / (1.f + expf(-gi));
  float sf = 1.f / (1.f + expf(-gf));
  float so = 1.f / (1.f + expf(-go));
  float cn = sf * c[idx] + si * tanhf(gg);
  float hn = so * tanhf(cn);
  h[idx] = m * hn + (1.f - m) * h[idx];
  c[idx] = m * cn + (1.f - m) * c[idx];
}

// out[n,o] = m * (h[n] . W_out[o] + b_out[o])  (h already masked-updated; equal when m==1)
__global__ void out_kernel(const float* __restrict__ h, const float* __restrict__ W_out,
                           const float* __restrict__ b_out, const int* __restrict__ pres_t,
                           float* __restrict__ out_t) {
  int idx = blockIdx.x * blockDim.x + threadIdx.x;
  if (idx >= NN * OUTD) return;
  int n = idx / OUTD, o = idx % OUTD;
  float s = b_out[o];
#pragma unroll 8
  for (int r = 0; r < RR; ++r) s += h[n * RR + r] * W_out[o * RR + r];
  out_t[idx] = s * (float)pres_t[n];
}

__global__ void finalize_kernel(const float* __restrict__ h, const float* __restrict__ c,
                                float* __restrict__ dst) {
  int i = blockIdx.x * blockDim.x + threadIdx.x;
  if (i < NN * RR) { dst[i] = h[i]; dst[NN * RR + i] = c[i]; }
}

// ---------------------------------- launcher ------------------------------------------

extern "C" void kernel_launch(void* const* d_in, const int* in_sizes, int n_in,
                              void* d_out, int out_size, void* d_ws, size_t ws_size,
                              hipStream_t stream) {
  const float* nodes   = (const float*)d_in[0];
  const float* grids   = (const float*)d_in[1];
  const int*   present = (const int*)  d_in[2];
  const float* h0      = (const float*)d_in[3];
  const float* c0      = (const float*)d_in[4];
  const float* W_in    = (const float*)d_in[5];
  const float* b_in    = (const float*)d_in[6];
  const float* W_ten   = (const float*)d_in[7];
  const float* b_ten   = (const float*)d_in[8];
  const float* W_ih    = (const float*)d_in[9];
  const float* W_hh    = (const float*)d_in[10];
  const float* b_ih    = (const float*)d_in[11];
  const float* b_hh    = (const float*)d_in[12];
  const float* W_out   = (const float*)d_in[13];
  const float* b_out   = (const float*)d_in[14];
  float* dout = (float*)d_out;

  char* base = (char*)d_ws;
  float*  h       = (float*)(base);                              // 256 KB
  float*  c       = (float*)(base + (size_t)256 * 1024);         // 256 KB
  float*  gates   = (float*)(base + (size_t)512 * 1024);         // 1 MB
  float*  tepart  = (float*)(base + (size_t)1536 * 1024);        // 2 MB
  __bf16* hmask   = (__bf16*)(base + (size_t)3584 * 1024);       // 128 KB
  __bf16* xcat    = (__bf16*)(base + (size_t)3712 * 1024);       // 256 KB
  __bf16* PbfT    = (__bf16*)(base + (size_t)3968 * 1024);       // 1 MB
  __bf16* wten_bf = (__bf16*)(base + (size_t)4992 * 1024);       // 256 KB
  __bf16* wcat_bf = (__bf16*)(base + (size_t)5248 * 1024);       // 256 KB (ends 5504 KB)

  init_state_kernel<<<(NN * RR + 255) / 256, 256, 0, stream>>>(h0, c0, h, c);
  convert_weights_kernel<<<(EE * G2 * RR + 255) / 256, 256, 0, stream>>>(
      W_ten, W_ih, W_hh, wten_bf, wcat_bf);

  for (int t = 0; t < SEQ; ++t) {
    const float* x_t     = nodes + (size_t)t * NN * 2;
    const float* grids_t = grids + (size_t)t * NN * KTEN;
    const int*   pres_t  = present + (size_t)t * NN;
    float*       out_t   = dout + (size_t)t * NN * OUTD;

    prep_kernel<<<NN, RR, 0, stream>>>(h, x_t, pres_t, W_in, b_in, hmask, xcat);
    p_kernel<<<64, 256, 0, stream>>>(hmask, wten_bf, PbfT);               // 512 waves
    te_kernel<<<64, 256, 0, stream>>>(grids_t, PbfT, tepart);             // 512 waves
    te_reduce_kernel<<<(NN * EE) / 256, 256, 0, stream>>>(tepart, b_ten, xcat);
    gates_kernel<<<32, 256, 0, stream>>>(xcat, wcat_bf, gates);           // 256 waves
    lstm_kernel<<<(NN * RR) / 256, 256, 0, stream>>>(gates, b_ih, b_hh, pres_t, h, c);
    out_kernel<<<(NN * OUTD + 255) / 256, 256, 0, stream>>>(h, W_out, b_out, pres_t, out_t);
  }

  finalize_kernel<<<(NN * RR + 255) / 256, 256, 0, stream>>>(h, c, dout + (size_t)SEQ * NN * OUTD);
}
